// DenseToSparseAttn_2413771621108
// MI455X (gfx1250) — matrix operand order repaired and checked by reading, not compiled, not run on hardware
//
#include <hip/hip_runtime.h>
#include <hip/hip_bf16.h>

#define EMB   512
#define NSTK  32
#define NPNT  64
#define BSZ   32
#define TILES_PER_BLOCK 4

typedef __attribute__((ext_vector_type(16))) __bf16 v16bf;
typedef __attribute__((ext_vector_type(8)))  float  v8f;

__device__ __forceinline__ unsigned short f2bf(float f) {
    unsigned int u = __float_as_uint(f);
    u += 0x7FFFu + ((u >> 16) & 1u);          // round-to-nearest-even
    return (unsigned short)(u >> 16);
}
__device__ __forceinline__ float leaky(float x) { return x > 0.0f ? x : 0.2f * x; }

// ---------------------------------------------------------------------------
// Kernel 1: convert Wk / Wg (fp32 512x512, row-major [o][c]) into bf16 in the
// WMMA A-fragment layout (16-bit A 16x32, ISA 7.12.2):
//   fragment f = mt*16 + kt ; element = lane*16 + e
//   m = mt*16 + (lane&15)
//   k = kt*32 + (e>>3)*16 + (lane>>4)*8 + (e&7)
// ---------------------------------------------------------------------------
__global__ void convert_weights_kernel(const float* __restrict__ Wk,
                                       const float* __restrict__ Wg,
                                       unsigned short* __restrict__ wkb,
                                       unsigned short* __restrict__ wgb) {
    unsigned int idx = blockIdx.x * blockDim.x + threadIdx.x;   // [0, 2*262144)
    const float* src = (idx < 262144u) ? Wk : Wg;
    unsigned short* dst = (idx < 262144u) ? wkb : wgb;
    unsigned int d    = idx & 262143u;
    unsigned int f    = d >> 9;
    unsigned int r    = d & 511u;
    unsigned int lane = r >> 4;
    unsigned int e    = r & 15u;
    unsigned int mt   = f >> 4, kt = f & 15u;
    unsigned int m = mt * 16u + (lane & 15u);
    unsigned int k = kt * 32u + (e >> 3) * 16u + (lane >> 4) * 8u + (e & 7u);
    dst[d] = f2bf(src[m * 512u + k]);
}

// ---------------------------------------------------------------------------
// Kernel 2: Q,V for all 1024 (b,s) columns.
//   q = leaky(Wq.x * sq + bq), v = leaky(Wv.x * sv + bv)
// grid (8 ochunks, 32 batches), 256 threads. Lane = stack index -> weight
// value broadcasts across all 32 lanes of a wave.
// ---------------------------------------------------------------------------
__global__ void qv_kernel(const float* __restrict__ sparse,
                          const float* __restrict__ Wq, const float* __restrict__ sq,
                          const float* __restrict__ bq,
                          const float* __restrict__ Wv, const float* __restrict__ sv,
                          const float* __restrict__ bv,
                          float* __restrict__ Q, float* __restrict__ V) {
    __shared__ float xs[EMB * NSTK];                  // 64 KB
    const int b  = blockIdx.y;
    const int oc = blockIdx.x;                        // 0..7
    const float* xsrc = sparse + b * EMB * NSTK;
    for (int i = threadIdx.x; i < EMB * NSTK; i += blockDim.x) xs[i] = xsrc[i];
    __syncthreads();

    const int s  = threadIdx.x & 31;
    const int og = threadIdx.x >> 5;                  // 0..7
    for (int ii = 0; ii < 8; ++ii) {
        const int o = oc * 64 + og * 8 + ii;
        const float* wqr = Wq + o * EMB;
        const float* wvr = Wv + o * EMB;
        float aq = 0.0f, av = 0.0f;
        for (int c = 0; c < EMB; ++c) {
            float x = xs[c * NSTK + s];
            aq = fmaf(wqr[c], x, aq);
            av = fmaf(wvr[c], x, av);
        }
        Q[(b * NSTK + s) * EMB + o] = leaky(aq * sq[o] + bq[o]);
        V[(b * NSTK + s) * EMB + o] = leaky(av * sv[o] + bv[o]);
    }
}

// ---------------------------------------------------------------------------
// B-fragment LDS helper: 16-bit B 32x16 fragment, frag fb = kt*4 + nt,
// lane lb : n = lb&15, half = lb>>4 ; element e : k_local = (e>>3)*16 + half*8 + (e&7).
// ---------------------------------------------------------------------------
__device__ __forceinline__ int bfrag_index(int k, int n) {
    int fb = (k >> 5) * 4 + (n >> 4);
    int lb = (n & 15) + (((k >> 3) & 1) << 4);
    int e  = (((k >> 4) & 1) << 3) | (k & 7);
    return fb * 512 + lb * 16 + e;
}

__device__ __forceinline__ void load_frags(v16bf a[2], v16bf bb[4],
                                           const unsigned short* __restrict__ wmat,
                                           const unsigned short* ldsB,
                                           int w, int lane, int kt) {
#pragma unroll
    for (int i = 0; i < 2; ++i)
        a[i] = *reinterpret_cast<const v16bf*>(wmat + (((w * 2 + i) * 16 + kt) * 512 + lane * 16));
#pragma unroll
    for (int j = 0; j < 4; ++j)
        bb[j] = *reinterpret_cast<const v16bf*>(&ldsB[(kt * 4 + j) * 512 + lane * 16]);
}

__device__ __forceinline__ void wmma8(v8f acc[2][4], const v16bf a[2], const v16bf bb[4]) {
#pragma unroll
    for (int i = 0; i < 2; ++i)
#pragma unroll
        for (int j = 0; j < 4; ++j)
            acc[i][j] = __builtin_amdgcn_wmma_f32_16x16x32_bf16(
                false, a[i], false, bb[j], (short)0, acc[i][j], false, false);
}

// One 512x64x512 bf16 GEMM sweep, software-pipelined: fragments for step kt+1
// are issued before the WMMAs of step kt so the s_wait_loadcnt/dscnt land a
// full stage later. Each wave owns 2 M-tiles x 4 N-tiles (32x64 of output).
__device__ __forceinline__ void gemm_sweep(v8f acc[2][4],
                                           const unsigned short* __restrict__ wmat,
                                           const unsigned short* ldsB,
                                           int w, int lane) {
    v16bf a[2], bb[4], an[2], bn[4];
    load_frags(a, bb, wmat, ldsB, w, lane, 0);
#pragma unroll 1
    for (int kt = 0; kt < 14; kt += 2) {
        load_frags(an, bn, wmat, ldsB, w, lane, kt + 1);
        wmma8(acc, a, bb);
        load_frags(a, bb, wmat, ldsB, w, lane, kt + 2);
        wmma8(acc, an, bn);
    }
    load_frags(an, bn, wmat, ldsB, w, lane, 15);
    wmma8(acc, a, bb);
    wmma8(acc, an, bn);
}

// ---------------------------------------------------------------------------
// Kernel 3: fused attention. Persistent blocks: 256 blocks x 4 (b,s) tiles,
// 512 threads = 16 wave32. While tile t computes its ~4096 WMMAs, the 128 KB
// dense slab of tile t+1 is prefetched into L2 (global_prefetch_b8), so the
// HBM stream of dense_fea overlaps matrix compute.
// ---------------------------------------------------------------------------
__global__ __launch_bounds__(512, 1)
void attn_kernel(const float* __restrict__ dense,
                 const unsigned short* __restrict__ wkb,
                 const unsigned short* __restrict__ wgb,
                 const float* __restrict__ sk, const float* __restrict__ bk,
                 const float* __restrict__ sg, const float* __restrict__ bg,
                 const float* __restrict__ Q,  const float* __restrict__ V,
                 float* __restrict__ out) {
    __shared__ __align__(32) unsigned short ldsB[EMB * NPNT];   // 64 KB, B-frag layout
    __shared__ float qsh[EMB];
    __shared__ float vsh[EMB];
    __shared__ float pmax[16][64];
    __shared__ float psum[16][64];
    __shared__ float colmax[64];
    __shared__ float colrcp[64];

    const int tid  = threadIdx.x;
    const int w    = tid >> 5;
    const int lane = tid & 31;
    const int half = lane >> 4;
    const int lan  = lane & 15;
    const v8f vzero = {0.f, 0.f, 0.f, 0.f, 0.f, 0.f, 0.f, 0.f};

#pragma unroll 1
    for (int tt = 0; tt < TILES_PER_BLOCK; ++tt) {
        const int bs = blockIdx.x * TILES_PER_BLOCK + tt;
        const int b  = bs >> 5;
        const int s  = bs & 31;

        // q, v columns for this (b,s)
        for (int i = tid; i < EMB; i += 512) {
            qsh[i] = Q[bs * EMB + i];
            vsh[i] = V[bs * EMB + i];
        }

        // D tile 512x64: coalesced fp32 load, bf16 swizzle into B-frag layout
        const float* dsrc = dense + (size_t)b * EMB * 2048 + (size_t)s * 64;
        for (int i = tid; i < EMB * NPNT; i += 512) {
            int k = i >> 6, n = i & 63;
            ldsB[bfrag_index(k, n)] = f2bf(dsrc[(size_t)k * 2048 + n]);
        }

        // warm L2 with the next tile's 128 KB (512 rows x 2 lines of 128 B)
        if (tt + 1 < TILES_PER_BLOCK) {
            const int nbs = bs + 1;
            const float* nsrc = dense + (size_t)(nbs >> 5) * EMB * 2048 + (size_t)(nbs & 31) * 64;
            for (int i = tid; i < 1024; i += 512)
                __builtin_prefetch(nsrc + (size_t)(i >> 1) * 2048 + (i & 1) * 32, 0, 1);
        }
        __syncthreads();

        v8f acc[2][4];
#pragma unroll
        for (int i = 0; i < 2; ++i)
#pragma unroll
            for (int j = 0; j < 4; ++j) acc[i][j] = vzero;

        // GEMM1: K = Wk . D
        gemm_sweep(acc, wkb, ldsB, w, lane);
        __syncthreads();                       // all waves done reading D tile

        // epilogue1: t = q - leaky(K*sk+bk), rewrite into ldsB (channel -> K)
#pragma unroll
        for (int i = 0; i < 2; ++i)
#pragma unroll
            for (int j = 0; j < 4; ++j)
#pragma unroll
                for (int r = 0; r < 8; ++r) {
                    int m = (w * 2 + i) * 16 + r + 8 * half;
                    int n = j * 16 + lan;
                    float hk = leaky(acc[i][j][r] * sk[m] + bk[m]);
                    ldsB[bfrag_index(m, n)] = f2bf(qsh[m] - hk);
                }
        __syncthreads();

        // GEMM2: G = Wg . T
#pragma unroll
        for (int i = 0; i < 2; ++i)
#pragma unroll
            for (int j = 0; j < 4; ++j) acc[i][j] = vzero;
        gemm_sweep(acc, wgb, ldsB, w, lane);

        // epilogue2: g = leaky(G*sg+bg)
#pragma unroll
        for (int i = 0; i < 2; ++i)
#pragma unroll
            for (int j = 0; j < 4; ++j)
#pragma unroll
                for (int r = 0; r < 8; ++r) {
                    int m = (w * 2 + i) * 16 + r + 8 * half;
                    acc[i][j][r] = leaky(acc[i][j][r] * sg[m] + bg[m]);
                }

        // ---- softmax over channel axis (512 rows) per column ----
#pragma unroll
        for (int j = 0; j < 4; ++j) {
            float mx = -1e30f;
#pragma unroll
            for (int i = 0; i < 2; ++i)
#pragma unroll
                for (int r = 0; r < 8; ++r) mx = fmaxf(mx, acc[i][j][r]);
            mx = fmaxf(mx, __shfl_xor(mx, 16, 32));     // pair the two row-halves
            if (half == 0) pmax[w][j * 16 + lan] = mx;
        }
        __syncthreads();
        if (tid < 64) {
            float m = pmax[0][tid];
#pragma unroll
            for (int ww = 1; ww < 16; ++ww) m = fmaxf(m, pmax[ww][tid]);
            colmax[tid] = m;
        }
        __syncthreads();

        // exp + column sums (exp values kept in the accumulators)
#pragma unroll
        for (int j = 0; j < 4; ++j) {
            float cm = colmax[j * 16 + lan];
            float ssum = 0.0f;
#pragma unroll
            for (int i = 0; i < 2; ++i)
#pragma unroll
                for (int r = 0; r < 8; ++r) {
                    float e = __expf(acc[i][j][r] - cm);
                    acc[i][j][r] = e;
                    ssum += e;
                }
            ssum += __shfl_xor(ssum, 16, 32);
            if (half == 0) psum[w][j * 16 + lan] = ssum;
        }
        __syncthreads();
        if (tid < 64) {
            float sm = 0.0f;
#pragma unroll
            for (int ww = 0; ww < 16; ++ww) sm += psum[ww][tid];
            colrcp[tid] = 1.0f / sm;
        }
        __syncthreads();

        // out[b,m,s] = v[m] * sum_p exp(g[m,p]-cm)/colsum[p]
#pragma unroll
        for (int i = 0; i < 2; ++i) {
            float rs[8] = {0.f, 0.f, 0.f, 0.f, 0.f, 0.f, 0.f, 0.f};
#pragma unroll
            for (int j = 0; j < 4; ++j) {
                float rc = colrcp[j * 16 + lan];
#pragma unroll
                for (int r = 0; r < 8; ++r) rs[r] += acc[i][j][r] * rc;
            }
#pragma unroll
            for (int r = 0; r < 8; ++r) {      // reduce over the 16 lanes of this half
                rs[r] += __shfl_xor(rs[r], 1, 32);
                rs[r] += __shfl_xor(rs[r], 2, 32);
                rs[r] += __shfl_xor(rs[r], 4, 32);
                rs[r] += __shfl_xor(rs[r], 8, 32);
            }
            if (lan == 0) {
#pragma unroll
                for (int r = 0; r < 8; ++r) {
                    int m = (w * 2 + i) * 16 + r + 8 * half;
                    out[((size_t)b * EMB + m) * NSTK + s] = vsh[m] * rs[r];
                }
            }
        }
        __syncthreads();            // protect qsh/vsh/ldsB before next tile
    }
}

// ---------------------------------------------------------------------------
extern "C" void kernel_launch(void* const* d_in, const int* in_sizes, int n_in,
                              void* d_out, int out_size, void* d_ws, size_t ws_size,
                              hipStream_t stream) {
    const float* sparse = (const float*)d_in[0];
    const float* dense  = (const float*)d_in[1];
    const float* Wq = (const float*)d_in[2];
    const float* sq = (const float*)d_in[3];
    const float* bq = (const float*)d_in[4];
    const float* Wk = (const float*)d_in[5];
    const float* sk = (const float*)d_in[6];
    const float* bk = (const float*)d_in[7];
    const float* Wv = (const float*)d_in[8];
    const float* sv = (const float*)d_in[9];
    const float* bv = (const float*)d_in[10];
    const float* Wg = (const float*)d_in[11];
    const float* sg = (const float*)d_in[12];
    const float* bg = (const float*)d_in[13];
    float* out = (float*)d_out;

    // workspace layout
    char* ws = (char*)d_ws;
    unsigned short* wkb = (unsigned short*)(ws);                 // 512 KB bf16 Wk fragments
    unsigned short* wgb = (unsigned short*)(ws + (512 << 10));   // 512 KB bf16 Wg fragments
    float* Q = (float*)(ws + (1 << 20));                         // 2 MB
    float* V = (float*)(ws + (3 << 20));                         // 2 MB

    convert_weights_kernel<<<2048, 256, 0, stream>>>(Wk, Wg, wkb, wgb);
    qv_kernel<<<dim3(8, BSZ), 256, 0, stream>>>(sparse, Wq, sq, bq, Wv, sv, bv, Q, V);
    attn_kernel<<<(BSZ * NSTK) / TILES_PER_BLOCK, 512, 0, stream>>>(
        dense, wkb, wgb, sk, bk, sg, bg, Q, V, out);
}
